// GAT_74423193305592
// MI455X (gfx1250) — matrix operand order repaired and checked by reading, not compiled, not run on hardware
//
#include <hip/hip_runtime.h>

// ---------- types for WMMA ----------
typedef __attribute__((ext_vector_type(16))) __bf16        v16bf;
typedef __attribute__((ext_vector_type(8)))  float         v8f;
typedef __attribute__((ext_vector_type(4)))  unsigned int  v4u;

union FragBF { v16bf v; v4u q[2]; };

// ---------- helpers ----------
__device__ __forceinline__ unsigned short f2bf(float f) {
    unsigned int u = __float_as_uint(f);
    unsigned int r = u + 0x7FFFu + ((u >> 16) & 1u);   // round-to-nearest-even
    return (unsigned short)(r >> 16);
}

__device__ __forceinline__ void atomicMaxF(float* addr, float val) {
    unsigned int* ua = (unsigned int*)addr;
    unsigned int cur = *ua;
    while (__uint_as_float(cur) < val) {
        unsigned int prev = atomicCAS(ua, cur, __float_as_uint(val));
        if (prev == cur) break;
        cur = prev;
    }
}

__device__ __forceinline__ void edge_ep(long e, const int* ei, int E, int& s, int& d) {
    if (e < E) { s = ei[e]; d = ei[(long)E + e]; }
    else       { s = d = (int)(e - E); }               // self-loop
}

__device__ __forceinline__ float leaky(float v) { return v > 0.f ? v : 0.2f * v; }

// ---------- generic utility kernels ----------
__global__ void k_fill(float* p, float v, long n) {
    long i = (long)blockIdx.x * blockDim.x + threadIdx.x;
    if (i < n) p[i] = v;
}

__global__ void k_f32_to_bf16(const float* in, unsigned short* out, long n) {
    long i = (long)blockIdx.x * blockDim.x + threadIdx.x;
    if (i < n) out[i] = f2bf(in[i]);
}

// W is [K][Nc] row-major f32 -> Wt is [Npad][K] bf16 (zero-fill padded cols)
__global__ void k_w_transpose_bf16(const float* W, unsigned short* Wt,
                                   int K, int Nc, int Npad) {
    long i = (long)blockIdx.x * blockDim.x + threadIdx.x;
    if (i >= (long)Npad * K) return;
    int n = (int)(i / K), k = (int)(i % K);
    float v = (n < Nc) ? W[(long)k * Nc + n] : 0.f;
    Wt[(long)n * K + k] = f2bf(v);
}

// ---------- WMMA GEMM: C[M x nmax] = A[M x K](bf16) * Bt[Npad x K]^T ----------
// one wave per 16x16 output tile; K multiple of 32
__global__ void __launch_bounds__(256)
k_gemm_bf16_wmma(const unsigned short* __restrict__ A,
                 const unsigned short* __restrict__ Bt,
                 float* __restrict__ C,
                 int M, int K, int ntiles, int ldc, int nmax) {
    int wave  = blockIdx.x * (blockDim.x >> 5) + (threadIdx.x >> 5);
    int lane  = threadIdx.x & 31;
    int mtiles = (M + 15) >> 4;
    if (wave >= mtiles * ntiles) return;               // wave-uniform exit (EXEC stays full)
    int mtile = wave / ntiles, ntile = wave % ntiles;

    int row = mtile * 16 + (lane & 15); if (row >= M) row = M - 1;
    int col = ntile * 16 + (lane & 15);
    // A fragment: lanes 0-15 hold K {0..7,16..23}, lanes 16-31 hold {8..15,24..31}
    const unsigned short* ap = A  + (long)row * K + ((lane >> 4) * 8);
    // B fragment: same K split as A (lanes 0-15 {0..7,16..23}, lanes 16-31 {8..15,24..31}) for its column (Bt transposed layout)
    const unsigned short* bp = Bt + (long)col * K + ((lane >> 4) * 8);

    v8f acc = {};
    for (int k0 = 0; k0 < K; k0 += 32) {
        FragBF a, b;
        a.q[0] = *(const v4u*)(ap + k0);
        a.q[1] = *(const v4u*)(ap + k0 + 16);
        b.q[0] = *(const v4u*)(bp + k0);
        b.q[1] = *(const v4u*)(bp + k0 + 16);
        acc = __builtin_amdgcn_wmma_f32_16x16x32_bf16(
                  false, a.v, false, b.v, (short)0, acc, false, false);
    }
    // C layout: VGPR r -> (M = r + 8*(lane>=16), N = lane&15)
    int mbase = mtile * 16 + (lane >> 4) * 8;
    int n = ntile * 16 + (lane & 15);
    if (n < nmax) {
        #pragma unroll
        for (int r = 0; r < 8; ++r) {
            int m = mbase + r;
            if (m < M) C[(long)m * ldc + n] = acc[r];
        }
    }
}

// ---------- attention score per node: a = xp . att ----------
__global__ void k_att(const float* __restrict__ xp,
                      const float* __restrict__ atts, const float* __restrict__ attd,
                      float* __restrict__ as, float* __restrict__ ad,
                      int N, int H, int Cc) {
    long i = (long)blockIdx.x * blockDim.x + threadIdx.x;
    if (i >= (long)N * H) return;
    int n = (int)(i / H), h = (int)(i % H);
    const float* xr = xp + (long)n * H * Cc + (long)h * Cc;
    float s = 0.f, d = 0.f;
    for (int c = 0; c < Cc; ++c) { float v = xr[c]; s += v * atts[h * Cc + c]; d += v * attd[h * Cc + c]; }
    as[i] = s; ad[i] = d;
}

// ---------- segment-softmax passes over edges ----------
__global__ void k_edge_max(const int* ei, int E, long Etot,
                           const float* as, const float* ad, float* mbuf, int H) {
    long i = (long)blockIdx.x * blockDim.x + threadIdx.x;
    if (i >= Etot * H) return;
    long e = i / H; int h = (int)(i % H);
    int s, d; edge_ep(e, ei, E, s, d);
    float v = leaky(as[(long)s * H + h] + ad[(long)d * H + h]);
    atomicMaxF(&mbuf[(long)d * H + h], v);
}

__global__ void k_edge_sum(const int* ei, int E, long Etot,
                           const float* as, const float* ad,
                           const float* mbuf, float* sbuf, int H) {
    long i = (long)blockIdx.x * blockDim.x + threadIdx.x;
    if (i >= Etot * H) return;
    long e = i / H; int h = (int)(i % H);
    int s, d; edge_ep(e, ei, E, s, d);
    float v = leaky(as[(long)s * H + h] + ad[(long)d * H + h]);
    atomicAdd(&sbuf[(long)d * H + h], __expf(v - mbuf[(long)d * H + h]));
}

__global__ void k_edge_agg(const int* ei, int E, long Etot,
                           const float* as, const float* ad,
                           const float* mbuf, const float* sbuf,
                           const float* __restrict__ xp, float* __restrict__ agg,
                           int H, int Cc) {
    long i = (long)blockIdx.x * blockDim.x + threadIdx.x;
    if (i >= Etot * H) return;
    long e = i / H; int h = (int)(i % H);
    int s, d; edge_ep(e, ei, E, s, d);
    float v = leaky(as[(long)s * H + h] + ad[(long)d * H + h]);
    float alpha = __expf(v - mbuf[(long)d * H + h]) / (sbuf[(long)d * H + h] + 1e-16f);
    const float* xr = xp  + (long)s * H * Cc + (long)h * Cc;
    float*       ar = agg + (long)d * H * Cc + (long)h * Cc;
    for (int c = 0; c < Cc; ++c) atomicAdd(&ar[c], alpha * xr[c]);
}

// ---------- ELU(+bias) -> bf16 for layer-2 GEMM input ----------
__global__ void k_elu_bf16(const float* agg, const float* b, unsigned short* xb,
                           long n, int D) {
    long i = (long)blockIdx.x * blockDim.x + threadIdx.x;
    if (i >= n) return;
    float v = agg[i] + b[i % D];
    v = v > 0.f ? v : (__expf(v) - 1.f);
    xb[i] = f2bf(v);
}

// ---------- final scores = agg2 + b2 ----------
__global__ void k_scores(const float* agg2, const float* b2, float* out_sc, long n, int Cc) {
    long i = (long)blockIdx.x * blockDim.x + threadIdx.x;
    if (i >= n) return;
    out_sc[i] = agg2[i] + b2[i % Cc];
}

// ---------- masked cross-entropy reduction + label copy ----------
__global__ void k_loss(const float* __restrict__ scores, const int* __restrict__ label,
                       const unsigned char* __restrict__ mask,
                       float* outLabel, float* red, int N, int Cc) {
    int n = blockIdx.x * blockDim.x + threadIdx.x;
    if (n >= N) return;
    const float* sc = scores + (long)n * Cc;
    float mx = -3.4e38f;
    for (int c = 0; c < Cc; ++c) mx = fmaxf(mx, sc[c]);
    float sum = 0.f;
    for (int c = 0; c < Cc; ++c) sum += __expf(sc[c] - mx);
    float lse = __logf(sum) + mx;
    int l = label[n];
    float nll = lse - sc[l];
    float w = mask[n] ? 1.f : 0.f;
    atomicAdd(&red[0], nll * w);
    atomicAdd(&red[1], w);
    outLabel[n] = (float)l;
}

__global__ void k_finalize(const float* red, float* out0) {
    if (threadIdx.x == 0 && blockIdx.x == 0) out0[0] = red[0] / red[1];
}

// =====================================================================
extern "C" void kernel_launch(void* const* d_in, const int* in_sizes, int n_in,
                              void* d_out, int out_size, void* d_ws, size_t ws_size,
                              hipStream_t stream) {
    const float*         feat  = (const float*)d_in[0];
    const int*           ei    = (const int*)d_in[1];
    const unsigned char* mask  = (const unsigned char*)d_in[2];
    const int*           label = (const int*)d_in[3];
    const float* W1  = (const float*)d_in[4];
    const float* as1 = (const float*)d_in[5];
    const float* ad1 = (const float*)d_in[6];
    const float* b1  = (const float*)d_in[7];
    const float* W2  = (const float*)d_in[8];
    const float* as2 = (const float*)d_in[9];
    const float* ad2 = (const float*)d_in[10];
    const float* b2  = (const float*)d_in[11];

    const int N    = in_sizes[2];            // 50000
    const int E    = in_sizes[1] / 2;        // 1.6M
    const int inD  = in_sizes[0] / N;        // 256
    const int H1   = 8;
    const int D1   = in_sizes[7];            // 256 = H1*hidden
    const int hid  = D1 / H1;                // 32
    const int C2   = in_sizes[11];           // 40
    const int C2p  = (C2 + 15) & ~15;        // 48
    const long Etot = (long)E + N;

    // ---- carve workspace ----
    char* p = (char*)d_ws;
    auto carve = [&](size_t bytes) -> void* {
        void* r = (void*)p; p += (bytes + 255) & ~(size_t)255; return r;
    };
    unsigned short* featb = (unsigned short*)carve((size_t)N * inD * 2);
    unsigned short* w1t   = (unsigned short*)carve((size_t)D1 * inD * 2);
    float* xp1   = (float*)carve((size_t)N * D1 * 4);
    float* asrc1 = (float*)carve((size_t)N * H1 * 4);
    float* adst1 = (float*)carve((size_t)N * H1 * 4);
    float* m1    = (float*)carve((size_t)N * H1 * 4);
    float* s1    = (float*)carve((size_t)N * H1 * 4);
    float* agg1  = (float*)carve((size_t)N * D1 * 4);
    unsigned short* xb2 = (unsigned short*)carve((size_t)N * D1 * 2);
    unsigned short* w2t = (unsigned short*)carve((size_t)C2p * D1 * 2);
    float* xp2   = (float*)carve((size_t)N * C2 * 4);
    float* asrc2 = (float*)carve((size_t)N * 4);
    float* adst2 = (float*)carve((size_t)N * 4);
    float* m2    = (float*)carve((size_t)N * 4);
    float* s2    = (float*)carve((size_t)N * 4);
    float* agg2  = (float*)carve((size_t)N * C2 * 4);
    float* red   = (float*)carve(2 * 4);

    float* out_loss   = (float*)d_out;
    float* out_scores = out_loss + 1;
    float* out_label  = out_scores + (long)N * C2;

    const int B = 256;
    auto nb = [&](long n) -> int { return (int)((n + B - 1) / B); };

    // ===== Layer 1 =====
    k_f32_to_bf16<<<nb((long)N * inD), B, 0, stream>>>(feat, featb, (long)N * inD);
    k_w_transpose_bf16<<<nb((long)D1 * inD), B, 0, stream>>>(W1, w1t, inD, D1, D1);

    {   // xp1 = feat @ W1   (WMMA bf16)
        int mt = (N + 15) / 16, nt = D1 / 16;
        int waves = mt * nt;
        k_gemm_bf16_wmma<<<(waves + 7) / 8, 256, 0, stream>>>(
            featb, w1t, xp1, N, inD, nt, D1, D1);
    }

    k_att<<<nb((long)N * H1), B, 0, stream>>>(xp1, as1, ad1, asrc1, adst1, N, H1, hid);

    k_fill<<<nb((long)N * H1), B, 0, stream>>>(m1, -3.4e38f, (long)N * H1);
    k_fill<<<nb((long)N * H1), B, 0, stream>>>(s1, 0.f, (long)N * H1);
    k_fill<<<nb((long)N * D1), B, 0, stream>>>(agg1, 0.f, (long)N * D1);

    k_edge_max<<<nb(Etot * H1), B, 0, stream>>>(ei, E, Etot, asrc1, adst1, m1, H1);
    k_edge_sum<<<nb(Etot * H1), B, 0, stream>>>(ei, E, Etot, asrc1, adst1, m1, s1, H1);
    k_edge_agg<<<nb(Etot * H1), B, 0, stream>>>(ei, E, Etot, asrc1, adst1, m1, s1,
                                                xp1, agg1, H1, hid);

    k_elu_bf16<<<nb((long)N * D1), B, 0, stream>>>(agg1, b1, xb2, (long)N * D1, D1);

    // ===== Layer 2 =====
    k_w_transpose_bf16<<<nb((long)C2p * D1), B, 0, stream>>>(W2, w2t, D1, C2, C2p);

    {   // xp2 = x @ W2   (WMMA bf16, N padded 40->48 with store guard)
        int mt = (N + 15) / 16, nt = C2p / 16;
        int waves = mt * nt;
        k_gemm_bf16_wmma<<<(waves + 7) / 8, 256, 0, stream>>>(
            xb2, w2t, xp2, N, D1, nt, C2, C2);
    }

    k_att<<<nb((long)N), B, 0, stream>>>(xp2, as2, ad2, asrc2, adst2, N, 1, C2);

    k_fill<<<nb((long)N), B, 0, stream>>>(m2, -3.4e38f, (long)N);
    k_fill<<<nb((long)N), B, 0, stream>>>(s2, 0.f, (long)N);
    k_fill<<<nb((long)N * C2), B, 0, stream>>>(agg2, 0.f, (long)N * C2);

    k_edge_max<<<nb(Etot), B, 0, stream>>>(ei, E, Etot, asrc2, adst2, m2, 1);
    k_edge_sum<<<nb(Etot), B, 0, stream>>>(ei, E, Etot, asrc2, adst2, m2, s2, 1);
    k_edge_agg<<<nb(Etot), B, 0, stream>>>(ei, E, Etot, asrc2, adst2, m2, s2,
                                           xp2, agg2, 1, C2);

    k_scores<<<nb((long)N * C2), B, 0, stream>>>(agg2, b2, out_scores, (long)N * C2, C2);

    // ===== loss =====
    k_fill<<<1, 2, 0, stream>>>(red, 0.f, 2);
    k_loss<<<nb((long)N), B, 0, stream>>>(out_scores, label, mask, out_label, red, N, C2);
    k_finalize<<<1, 1, 0, stream>>>(red, out_loss);
}
